// ScalarDotProductAttention_26147760898451
// MI455X (gfx1250) — matrix operand order repaired and checked by reading, not compiled, not run on hardware
//
#include <hip/hip_runtime.h>
#include <hip/hip_bf16.h>
#include <math.h>

typedef __attribute__((ext_vector_type(16))) _Float16 v16h;
typedef __attribute__((ext_vector_type(8)))  _Float16 v8h;
typedef __attribute__((ext_vector_type(2)))  _Float16 v2h;
typedef __attribute__((ext_vector_type(8)))  float    v8f;
typedef __attribute__((ext_vector_type(4)))  float    v4f;

#define WMMA_F16(a, b, c) \
  __builtin_amdgcn_wmma_f32_16x16x32_f16(false, (a), false, (b), (short)0, (c), false, false)

constexpr int L_SEQ = 2048, H_HEADS = 16, D_HEAD = 64;
constexpr int BM  = 64;       // queries per workgroup (16 per wave, 4 waves)
constexpr int BN  = 128;      // keys per iteration
constexpr int NT  = BN / 16;  // 8 score tiles
constexpr int KC  = BN / 32;  // 4 A-chunks for P*V
constexpr int LDK = 72;       // sK row pitch (f16)
constexpr int LDV = 136;      // sVt / sP row pitch (f16)

// Finite sentinel far below any achievable log2-domain score;
// exp2(sentinel - m) underflows cleanly to 0 for the first block's alpha.
constexpr float NEG_SENTINEL = -1.0e4f;

// Raw v_exp_f32: exact for args <= 0; denorm flush is fine for softmax.
__device__ __forceinline__ float fexp2(float x) { return __builtin_amdgcn_exp2f(x); }

// max(a,b) via v_med3_num_f32 with +FLT_MAX: single instruction, no operand
// canonicalization, and (being compiler-visible) gets correct WMMA->VALU
// hazard NOPs — unlike an opaque inline-asm v_max.
__device__ __forceinline__ float vmax(float a, float b) {
  return __builtin_amdgcn_fmed3f(a, b, 3.402823466e+38f);
}

// XOR-butterfly shuffle within 32 lanes via ds_swizzle (group-of-32 mode)
template <int MASK>
__device__ __forceinline__ float xor_shfl(float v) {
  return __int_as_float(
      __builtin_amdgcn_ds_swizzle(__float_as_int(v), (MASK << 10) | 0x1f));
}

__launch_bounds__(128)
__global__ void fa_fwd_kernel(const float* __restrict__ Q,
                              const float* __restrict__ Km,
                              const float* __restrict__ V,
                              float* __restrict__ Out)
{
  __shared__ _Float16 sK [BN][LDK];        // K block, [key][d], f16
  __shared__ _Float16 sVt[D_HEAD][LDV];    // V block transposed, [d][key], f16
  __shared__ _Float16 sP [4][16][LDV];     // per-wave P bounce, [wave][row][key]

  const int tid  = threadIdx.x;
  const int wid  = tid >> 5;
  const int lane = tid & 31;
  const int l16  = lane & 15;
  const int hi   = lane >> 4;              // half-wave id

  const int qblk = blockIdx.x;             // 0..31
  const int h    = blockIdx.y;             // 0..15
  const int n    = blockIdx.z;             // 0..1

  const size_t headoff = (size_t)n * L_SEQ * H_HEADS * D_HEAD + (size_t)h * D_HEAD;
  const float* Qb = Q  + headoff;
  const float* Kb = Km + headoff;
  const float* Vb = V  + headoff;
  float*       Ob = Out + headoff;
  const int rowstride = H_HEADS * D_HEAD;  // 1024 floats between seq rows

  // softmax scale folded into Q: x = dot * (1/sqrt(D)) * log2(e)
  const float kScl = 0.125f * 1.44269504088896340736f;

  // ---- preload this wave's Q tile in WMMA A-layout (16 rows x 64 d) ----
  const int qrow = qblk * BM + wid * 16 + l16;
  const float* qp = Qb + (size_t)qrow * rowstride;
  v16h aQ[2];
  #pragma unroll
  for (int c = 0; c < 2; ++c)
    #pragma unroll
    for (int j = 0; j < 16; ++j) {
      // A k order chosen here (B matches): element j <-> K = 16*hi + j
      int d = 32 * c + 16 * hi + j;
      aQ[c][j] = (_Float16)(qp[d] * kScl);
    }

  v16h vones;
  #pragma unroll
  for (int j = 0; j < 16; ++j) vones[j] = (_Float16)1.0f;

  v8f acc[4] = {};                          // O accumulators (4 d-chunks of 16)
  v8f lacc = {};                            // row-sum accumulator (P x ones)
  float m_i[8];
  #pragma unroll
  for (int r = 0; r < 8; ++r) m_i[r] = NEG_SENTINEL;

  for (int kb = 0; kb < L_SEQ; kb += BN) {
    __syncthreads();
    // ---- stage K -> sK (f16): one key row per thread, packed b128 stores ----
    {
      const int r = tid;                    // 0..127 : key row
      const float* kp = Kb + (size_t)(kb + r) * rowstride;
      #pragma unroll
      for (int j = 0; j < 64; j += 8) {
        v4f a = *(const v4f*)(kp + j);
        v4f b = *(const v4f*)(kp + j + 4);
        v8h p;
        #pragma unroll
        for (int e = 0; e < 4; ++e) { p[e] = (_Float16)a[e]; p[4 + e] = (_Float16)b[e]; }
        *(v8h*)&sK[r][j] = p;
      }
    }
    // ---- stage V -> sVt (f16, transposed): 2 keys x 32 d per thread ----
    {
      const int k0 = (tid & 63) * 2;
      const int d0 = (tid >> 6) * 32;
      const float* v0 = Vb + (size_t)(kb + k0) * rowstride + d0;
      const float* v1 = v0 + rowstride;
      #pragma unroll
      for (int j = 0; j < 32; j += 4) {
        v4f x = *(const v4f*)(v0 + j);
        v4f y = *(const v4f*)(v1 + j);
        #pragma unroll
        for (int e = 0; e < 4; ++e) {
          v2h p; p[0] = (_Float16)x[e]; p[1] = (_Float16)y[e];
          *(v2h*)&sVt[d0 + j + e][k0] = p;     // pairs along k -> b32 store
        }
      }
    }
    __syncthreads();

    // ---- prefetch next K/V block into cache while we compute ----
    if (kb + BN < L_SEQ) {
      const float* nk = Kb + (size_t)(kb + BN + tid) * rowstride;
      __builtin_prefetch(nk,      0, 1);
      __builtin_prefetch(nk + 32, 0, 1);
      const float* nv = Vb + (size_t)(kb + BN + (tid & 63) * 2) * rowstride + (tid >> 6) * 32;
      __builtin_prefetch(nv,             0, 1);
      __builtin_prefetch(nv + rowstride, 0, 1);
    }

    // ---- S = Q * K^T (already in log2 domain): 8 tiles x 2 WMMAs ----
    v8f S[NT];
    #pragma unroll
    for (int t = 0; t < NT; ++t) {
      v16h b0, b1;
      #pragma unroll
      for (int j = 0; j < 16; ++j) {
        // B: lane -> N = l16, element j -> K = 16*hi + j (same k order as A)
        int key = 16 * t + l16;
        b0[j] = sK[key][ 0 + 16 * hi + j];
        b1[j] = sK[key][32 + 16 * hi + j];
      }
      v8f z = {};
      z = WMMA_F16(aQ[0], b0, z);
      S[t] = WMMA_F16(aQ[1], b1, z);
    }

    // ---- online softmax: row max + rescale ----
    float mnew[8], alpha[8];
    #pragma unroll
    for (int r = 0; r < 8; ++r) {
      float v = S[0][r];
      #pragma unroll
      for (int t = 1; t < NT; ++t) v = vmax(v, S[t][r]);
      v = vmax(v, xor_shfl<1>(v));
      v = vmax(v, xor_shfl<2>(v));
      v = vmax(v, xor_shfl<4>(v));
      v = vmax(v, xor_shfl<8>(v));           // row max within 16-lane half
      mnew[r]  = vmax(m_i[r], v);
      alpha[r] = fexp2(m_i[r] - mnew[r]);
      m_i[r]   = mnew[r];
    }
    #pragma unroll
    for (int t = 0; t < NT; ++t)
      #pragma unroll
      for (int r = 0; r < 8; ++r)
        S[t][r] = fexp2(S[t][r] - mnew[r]);
    #pragma unroll
    for (int dt = 0; dt < 4; ++dt)
      #pragma unroll
      for (int r = 0; r < 8; ++r)
        acc[dt][r] *= alpha[r];
    #pragma unroll
    for (int r = 0; r < 8; ++r) lacc[r] *= alpha[r];

    // ---- P (C-layout f32) -> per-wave LDS f16 -> reload in A-layout ----
    #pragma unroll
    for (int t = 0; t < NT; ++t)
      #pragma unroll
      for (int r = 0; r < 8; ++r)
        sP[wid][r + 8 * hi][16 * t + l16] = (_Float16)S[t][r];
    asm volatile("s_wait_dscnt 0" ::: "memory");  // same-wave LDS RAW ordering

    v16h aP[KC];
    #pragma unroll
    for (int kc = 0; kc < KC; ++kc)
      #pragma unroll
      for (int j = 0; j < 16; ++j) {
        int key = 32 * kc + 16 * hi + j;
        aP[kc][j] = sP[wid][l16][key];
      }

    // ---- row sums via WMMA: lacc += P x ones ----
    #pragma unroll
    for (int kc = 0; kc < KC; ++kc)
      lacc = WMMA_F16(aP[kc], vones, lacc);

    // ---- O += P * V : 4 d-tiles x 4 key chunks ----
    #pragma unroll
    for (int dt = 0; dt < 4; ++dt) {
      #pragma unroll
      for (int kc = 0; kc < KC; ++kc) {
        v16h bv;
        #pragma unroll
        for (int j = 0; j < 16; ++j)
          bv[j] = sVt[16 * dt + l16][32 * kc + 16 * hi + j];
        acc[dt] = WMMA_F16(aP[kc], bv, acc[dt]);
      }
    }
  }

  // ---- epilogue: normalize by row sums and store fp32 ----
  #pragma unroll
  for (int r = 0; r < 8; ++r) {
    float inv = __builtin_amdgcn_rcpf(lacc[r]);
    int q = qblk * BM + wid * 16 + r + 8 * hi;
    float* op = Ob + (size_t)q * rowstride;
    #pragma unroll
    for (int dt = 0; dt < 4; ++dt)
      op[16 * dt + l16] = acc[dt][r] * inv;
  }
}

extern "C" void kernel_launch(void* const* d_in, const int* in_sizes, int n_in,
                              void* d_out, int out_size, void* d_ws, size_t ws_size,
                              hipStream_t stream) {
  (void)in_sizes; (void)n_in; (void)out_size; (void)d_ws; (void)ws_size;
  const float* Q = (const float*)d_in[0];
  const float* K = (const float*)d_in[1];
  const float* V = (const float*)d_in[2];
  float* Out = (float*)d_out;

  dim3 grid(L_SEQ / BM, H_HEADS, 2);   // (query blocks, heads, batch)
  dim3 block(128);                     // 4 waves of 32
  fa_fwd_kernel<<<grid, block, 0, stream>>>(Q, K, V, Out);
}
